// Autocorrelation_43834436223080
// MI455X (gfx1250) — matrix operand order repaired and checked by reading, not compiled, not run on hardware
//
#include <hip/hip_runtime.h>

// ---------------------------------------------------------------------------
// AutoCorrelation attention for MI455X (gfx1250).
// B=32, T=2048, D=512, TOPK=15.
// GEMMs: v_wmma_f32_16x16x32_bf16 (f32 accumulate), b128 LDS fragment loads.
// Correlation: packed-real radix-2 FFT in LDS; channel data staged by the
// Tensor Data Mover (tensor_load_to_lds + s_wait_tensorcnt).
// ---------------------------------------------------------------------------

#define B_  32
#define T_  2048
#define D_  512
#define TOPK_ 15

typedef __bf16 bf16_t;
typedef bf16_t v16bf __attribute__((ext_vector_type(16)));
typedef float  v8f   __attribute__((ext_vector_type(8)));
typedef unsigned int u32x4 __attribute__((ext_vector_type(4)));
typedef int          i32x8 __attribute__((ext_vector_type(8)));
typedef int          i32x4 __attribute__((ext_vector_type(4)));

union FragBF { v16bf v; unsigned short u[16]; uint4 q[2]; };

__device__ inline unsigned short f2bf(float f) {
  unsigned int u = __float_as_uint(f);
  return (unsigned short)((u + 0x7FFFu + ((u >> 16) & 1u)) >> 16);  // RNE
}
__device__ inline unsigned short ld_bf(const float* p)          { return f2bf(*p); }
__device__ inline unsigned short ld_bf(const unsigned short* p) { return *p; }

// ---------------------------------------------------------------------------
// Tensor Data Mover: contiguous 2048-float (8 KB) global -> LDS copy.
// D# built per CDNA5 ISA ch.8 (group0/group1 bitfields), 2D degenerate tile.
// ---------------------------------------------------------------------------
__device__ inline void tdm_load_2048f(const float* gsrc, void* lds_dst) {
  unsigned long long ga = (unsigned long long)gsrc;
  unsigned lds = (unsigned)(unsigned long long)lds_dst;  // LDS aperture low bits
  u32x4 g0;
  g0[0] = 1u;                                            // count=1 valid user D#
  g0[1] = lds;                                           // lds_addr
  g0[2] = (unsigned)(ga & 0xFFFFFFFFu);                  // global_addr[31:0]
  g0[3] = (unsigned)((ga >> 32) & 0x01FFFFFFu) | (2u << 30);  // [120:64], type=2
  i32x8 g1;
  g1[0] = (int)(2u << 16);        // data_size=4B; no mask/pad/iterate
  g1[1] = (int)0x08000000u;       // tensor_dim0 = 2048 (bits [79:48] low half)
  g1[2] = (int)0x00010000u;       // tensor_dim0 hi=0 | tensor_dim1 = 1
  g1[3] = (int)0x08000000u;       // tensor_dim1 hi=0 | tile_dim0 = 2048
  g1[4] = 1;                      // tile_dim1 = 1, tile_dim2 = 0
  g1[5] = 2048;                   // tensor_dim0_stride[31:0]
  g1[6] = (int)0x08000000u;       // stride0 hi=0 | tensor_dim1_stride lo = 2048
  g1[7] = 0;
  i32x4 z4 = {0, 0, 0, 0};
#if __clang_major__ >= 23
  i32x8 z8 = {0, 0, 0, 0, 0, 0, 0, 0};
  __builtin_amdgcn_tensor_load_to_lds(g0, g1, z4, z4, z8, 0);
#else
  __builtin_amdgcn_tensor_load_to_lds(g0, g1, z4, z4, 0);
#endif
}

// ---------------------------------------------------------------------------
// Tiled GEMM: out[M,512] = X[M,512] @ W[512,512] + bias, bf16 WMMA, f32 accum.
// LDS: As row-major [128][32], Bs n-major [128 n][32 k] so every fragment is
// contiguous -> ds_load_b128 only. 256 threads = 8 waves (4x2 of 32x64).
// TRANS=true -> out is [B, D, T] via LDS transpose epilogue.
// ---------------------------------------------------------------------------
template <typename InT, bool TRANS>
__global__ __launch_bounds__(256) void gemm_bias(const InT* X, const float* W,
                                                 const float* bias, float* out) {
  constexpr int SMEM_FLOATS = TRANS ? (128 * 129) : 4096;
  __shared__ float smem[SMEM_FLOATS];
  unsigned short* As = (unsigned short*)smem;   // [128][32] bf16 bits (8 KB)
  unsigned short* Bs = As + 128 * 32;           // [128 n][32 k] bf16 bits (8 KB)
  float* Cs = smem;                             // reused for transpose epilogue

  const int tid  = threadIdx.x;
  const int lane = tid & 31;
  const int half = lane >> 4;     // wave32 half select per ISA layouts
  const int lr   = lane & 15;
  const int wave = tid >> 5;
  const int wm   = wave >> 1;     // 0..3 -> 32-row strip
  const int wn   = wave & 1;      // 0..1 -> 64-col strip

  const int row0 = blockIdx.x * 128;
  const int n0   = blockIdx.y * 128;

  v8f acc[2][4];
#pragma unroll
  for (int fm = 0; fm < 2; ++fm)
#pragma unroll
    for (int fn = 0; fn < 4; ++fn)
#pragma unroll
      for (int i = 0; i < 8; ++i) acc[fm][fn][i] = 0.f;

  for (int k0 = 0; k0 < 512; k0 += 32) {
    // --- A tile: 128x32, coalesced global, b64 LDS stores ---
#pragma unroll
    for (int i = 0; i < 4; ++i) {
      int r  = (tid >> 3) + i * 32;
      int kk = (tid & 7) * 4;
      const InT* src = X + (size_t)(row0 + r) * 512 + k0 + kk;
      uint2 t2;
      t2.x = (unsigned)ld_bf(src)     | ((unsigned)ld_bf(src + 1) << 16);
      t2.y = (unsigned)ld_bf(src + 2) | ((unsigned)ld_bf(src + 3) << 16);
      *(uint2*)(As + r * 32 + kk) = t2;
    }
    // --- B tile: lanes coalesce across n, each thread packs 4 k (b64 store) ---
    {
      int nn  = tid & 31;
      int kk4 = (tid >> 5) * 4;
#pragma unroll
      for (int ni = 0; ni < 4; ++ni) {
        int n = nn + ni * 32;
        const float* wp = W + (size_t)(k0 + kk4) * 512 + n0 + n;
        uint2 t2;
        t2.x = (unsigned)f2bf(wp[0])    | ((unsigned)f2bf(wp[512])  << 16);
        t2.y = (unsigned)f2bf(wp[1024]) | ((unsigned)f2bf(wp[1536]) << 16);
        *(uint2*)(Bs + n * 32 + kk4) = t2;
      }
    }
    __syncthreads();

    // --- fragments: contiguous 16B runs -> ds_load_b128 ---
    FragBF fa[2], fb[4];
#pragma unroll
    for (int fm = 0; fm < 2; ++fm) {
      int ar = wm * 32 + fm * 16 + lr;
      fa[fm].q[0] = *(const uint4*)(As + ar * 32 + half * 8);       // K h8..h8+7
      fa[fm].q[1] = *(const uint4*)(As + ar * 32 + 16 + half * 8);  // K 16+h8..
    }
#pragma unroll
    for (int fn = 0; fn < 4; ++fn) {
      int bc = wn * 64 + fn * 16 + lr;
      fb[fn].q[0] = *(const uint4*)(Bs + bc * 32 + half * 8);       // K h8..h8+7
      fb[fn].q[1] = *(const uint4*)(Bs + bc * 32 + 16 + half * 8);  // K 16+h8..
    }

#pragma unroll
    for (int fm = 0; fm < 2; ++fm)
#pragma unroll
      for (int fn = 0; fn < 4; ++fn)
        acc[fm][fn] = __builtin_amdgcn_wmma_f32_16x16x32_bf16(
            false, fa[fm].v, false, fb[fn].v, (short)0, acc[fm][fn],
            false, false);
    __syncthreads();
  }

  if (!TRANS) {
    // direct store: C layout m = r + 8*half, n = lane&15 (coalesced across lr)
#pragma unroll
    for (int fm = 0; fm < 2; ++fm)
#pragma unroll
      for (int fn = 0; fn < 4; ++fn) {
        int mg = row0 + wm * 32 + fm * 16 + 8 * half;
        int ng = n0 + wn * 64 + fn * 16 + lr;
        float bv = bias[ng];
#pragma unroll
        for (int r = 0; r < 8; ++r)
          out[(size_t)(mg + r) * 512 + ng] = acc[fm][fn][r] + bv;
      }
  } else {
    // transpose through LDS so [B,D,T] stores are coalesced along T
#pragma unroll
    for (int fm = 0; fm < 2; ++fm)
#pragma unroll
      for (int fn = 0; fn < 4; ++fn) {
        int ml = wm * 32 + fm * 16 + 8 * half;
        int nl = wn * 64 + fn * 16 + lr;
        float bv = bias[n0 + nl];
#pragma unroll
        for (int r = 0; r < 8; ++r)
          Cs[(ml + r) * 129 + nl] = acc[fm][fn][r] + bv;
      }
    __syncthreads();
    const int bidx = row0 >> 11;        // T=2048 rows per batch
    const int tb   = row0 & 2047;
    float* ob = out + (size_t)bidx * D_ * T_;
#pragma unroll 4
    for (int p = 0; p < 64; ++p) {
      int flat = p * 256 + tid;
      int tl = flat & 127, nr = flat >> 7;
      ob[(size_t)(n0 + nr) * T_ + tb + tl] = Cs[tl * 129 + nr];
    }
  }
}

// ---------------------------------------------------------------------------
// Packed-real FFT correlation. One workgroup per (b,d) channel.
// z = q + i*k ; Z = FFT(z); unpack Qf,Kf; W = Qf*conj(Kf); attn = Re IFFT(W).
// attn aliases q_t (each block only touches its own channel).
// ---------------------------------------------------------------------------
__device__ inline void fft_stages(float* re, float* im, int tid, float sign) {
  for (int s = 0; s < 11; ++s) {
    int h = 1 << s;
    for (int u = tid; u < 1024; u += 256) {
      int j  = u & (h - 1);
      int i0 = ((u >> s) << (s + 1)) + j;
      int i1 = i0 + h;
      float ang = sign * 3.14159265358979f * (float)j / (float)h;
      float sn = __sinf(ang), cs = __cosf(ang);
      float xr = re[i1], xi = im[i1];
      float tr = cs * xr - sn * xi;
      float ti = cs * xi + sn * xr;
      float yr = re[i0], yi = im[i0];
      re[i1] = yr - tr; im[i1] = yi - ti;
      re[i0] = yr + tr; im[i0] = yi + ti;
    }
    __syncthreads();
  }
}

__device__ inline void bitrev2048(float* re, float* im, int tid) {
  for (int i = tid; i < 2048; i += 256) {
    int j = (int)(__brev((unsigned)i) >> 21);
    if (j > i) {
      float tr = re[i]; re[i] = re[j]; re[j] = tr;
      float ti = im[i]; im[i] = im[j]; im[j] = ti;
    }
  }
  __syncthreads();
}

__global__ __launch_bounds__(256) void fft_corr(const float* qt, const float* kt,
                                                float* attn) {
  __shared__ float re[2048];
  __shared__ float im[2048];
  const int tid = threadIdx.x;
  const size_t base = (size_t)blockIdx.x * 2048;

  // TDM DMA of both 8 KB channels into LDS; wave 0 issues (TDM ignores EXEC,
  // gate by control flow), waits TENSORcnt, then the group barrier releases.
  if (tid < 32) {
    tdm_load_2048f(qt + base, (void*)re);
    tdm_load_2048f(kt + base, (void*)im);
    __builtin_amdgcn_s_wait_tensorcnt(0);
  }
  __syncthreads();

  bitrev2048(re, im, tid);
  fft_stages(re, im, tid, -1.f);          // forward

  // Hermitian unpack + spectral multiply; thread f owns pair {f, 2048-f}
  for (int f = tid; f <= 1024; f += 256) {
    int g = (2048 - f) & 2047;
    float ar = re[f], ai = im[f];
    float br = re[g], bi = -im[g];
    float qr = 0.5f * (ar + br), qi = 0.5f * (ai + bi);      // Qf
    float dr = ar - br,          di = ai - bi;
    float kr = 0.5f * di,        ki = -0.5f * dr;            // Kf
    float wr = qr * kr + qi * ki;                            // Qf*conj(Kf)
    float wi = qi * kr - qr * ki;
    re[f] = wr; im[f] = wi;
    if (g != f) { re[g] = wr; im[g] = -wi; }
  }
  __syncthreads();

  bitrev2048(re, im, tid);
  fft_stages(re, im, tid, +1.f);          // inverse (scale below)

  const float sc = 1.f / 2048.f;
  for (int i = tid; i < 2048; i += 256) attn[base + i] = re[i] * sc;
}

// ---------------------------------------------------------------------------
// mean over d: attn is [B, D, T]; coalesced along t.
// ---------------------------------------------------------------------------
__global__ __launch_bounds__(256) void mean_kernel(const float* attn, float* mean) {
  int t = blockIdx.x * 256 + threadIdx.x;
  int b = blockIdx.y;
  const float* p = attn + (size_t)b * D_ * T_ + t;
  float s = 0.f;
  for (int d = 0; d < D_; ++d) s += p[(size_t)d * T_];
  mean[b * T_ + t] = s * (1.f / (float)D_);
}

// ---------------------------------------------------------------------------
// top-15 per batch: 15 masked argmax reductions (deterministic; lower index
// wins ties, matching lax.top_k ordering).
// ---------------------------------------------------------------------------
__global__ __launch_bounds__(256) void topk_kernel(const float* mean, int* delays) {
  __shared__ float sv[2048];
  __shared__ float rv[256];
  __shared__ int   ri[256];
  const int tid = threadIdx.x, b = blockIdx.x;
  for (int i = tid; i < 2048; i += 256) sv[i] = mean[b * T_ + i];
  __syncthreads();
  for (int k = 0; k < TOPK_; ++k) {
    float bv = -3.4e38f; int bi = 2048;
    for (int i = tid; i < 2048; i += 256) {
      float v = sv[i];
      if (v > bv) { bv = v; bi = i; }
    }
    rv[tid] = bv; ri[tid] = bi;
    __syncthreads();
    for (int off = 128; off > 0; off >>= 1) {
      if (tid < off) {
        float ov = rv[tid + off]; int oi = ri[tid + off];
        if (ov > rv[tid] || (ov == rv[tid] && oi < ri[tid])) { rv[tid] = ov; ri[tid] = oi; }
      }
      __syncthreads();
    }
    if (tid == 0) { delays[b * TOPK_ + k] = ri[0]; sv[ri[0]] = -3.4e38f; }
    __syncthreads();
  }
}

// ---------------------------------------------------------------------------
// Fused: softmax over d of attn[b,t,:] (online, via 512x32 LDS transpose tile)
// times rolled-sum of v over the 15 delays; emits bf16 for the final GEMM.
// attn is [B,D,T]; v is [B,T,D]; out xbf is bf16 [B,T,D].
// ---------------------------------------------------------------------------
__global__ __launch_bounds__(256) void softroll_kernel(const float* attn,
                                                       const float* v,
                                                       const int* delays,
                                                       unsigned short* xbf) {
  __shared__ float tile[512 * 36];      // [d][t] transposed block (pitch 36)
  __shared__ float pm[8 * 32];
  __shared__ float ps[8 * 32];
  __shared__ float Mx[32];
  __shared__ float Sinv[32];
  __shared__ int   dly[TOPK_];

  const int tid = threadIdx.x;
  const int b = blockIdx.y, t0 = blockIdx.x * 32;
  if (tid < TOPK_) dly[tid] = delays[b * TOPK_ + tid];

  const float* ab = attn + (size_t)b * D_ * T_;
#pragma unroll 4
  for (int p = 0; p < 16; ++p) {        // 512 rows x 32 t, float4 per lane
    int e4 = (p * 256 + tid) * 4;
    int d = e4 >> 5, tt = e4 & 31;
    *(float4*)(tile + d * 36 + tt) = *(const float4*)(ab + (size_t)d * T_ + t0 + tt);
  }
  __syncthreads();

  {  // online (max,sum) partials: 8 d-lanes x 32 t
    int t = tid & 31, dl = tid >> 5;
    float m = -3.4e38f, s = 0.f;
    for (int i = 0; i < 64; ++i) {
      float x = tile[(dl * 64 + i) * 36 + t];
      float nm = fmaxf(m, x);
      s = s * __expf(m - nm) + __expf(x - nm);
      m = nm;
    }
    pm[dl * 32 + t] = m; ps[dl * 32 + t] = s;
  }
  __syncthreads();
  if (tid < 32) {
    float m = pm[tid], s = ps[tid];
    for (int l = 1; l < 8; ++l) {
      float m2 = pm[l * 32 + tid], s2 = ps[l * 32 + tid];
      float nm = fmaxf(m, m2);
      s = s * __expf(m - nm) + s2 * __expf(m2 - nm);
      m = nm;
    }
    Mx[tid] = m; Sinv[tid] = 1.f / s;
  }
  __syncthreads();

  const int dcol = tid & 31, trow = tid >> 5;
  const float* vb = v + (size_t)b * T_ * D_;
  unsigned short* ob = xbf + (size_t)b * T_ * D_;
  for (int ti = 0; ti < 4; ++ti) {
    int tl = trow + ti * 8;
    int tg = t0 + tl;
    int rt[TOPK_];
#pragma unroll
    for (int k = 0; k < TOPK_; ++k) rt[k] = (tg - dly[k]) & (T_ - 1);
    float mm = Mx[tl], si = Sinv[tl];
    for (int d0 = 0; d0 < D_; d0 += 32) {
      int d = d0 + dcol;
      float acc = 0.f;
#pragma unroll
      for (int k = 0; k < TOPK_; ++k) acc += vb[(size_t)rt[k] * D_ + d];  // L2-resident
      float pr = __expf(tile[d * 36 + tl] - mm) * si;
      ob[(size_t)tg * D_ + d] = f2bf(acc * pr);
    }
  }
}

// ---------------------------------------------------------------------------
extern "C" void kernel_launch(void* const* d_in, const int* in_sizes, int n_in,
                              void* d_out, int out_size, void* d_ws, size_t ws_size,
                              hipStream_t stream) {
  const float* query  = (const float*)d_in[0];
  const float* key_in = (const float*)d_in[1];
  const float* value  = (const float*)d_in[2];
  const float* Wq = (const float*)d_in[3];  const float* bq = (const float*)d_in[4];
  const float* Wk = (const float*)d_in[5];  const float* bk = (const float*)d_in[6];
  const float* Wv = (const float*)d_in[7];  const float* bv = (const float*)d_in[8];
  const float* Wo = (const float*)d_in[9];  const float* bo = (const float*)d_in[10];
  float* out = (float*)d_out;

  char* ws = (char*)d_ws;
  const size_t TENS = (size_t)B_ * T_ * D_ * sizeof(float);   // 134 MB
  float* q_t  = (float*)(ws);                  // [B,D,T]; later aliased by attn
  float* k_t  = (float*)(ws + TENS);           // [B,D,T]; later aliased by xbf
  float* v_   = (float*)(ws + 2 * TENS);       // [B,T,D]
  float* mean = (float*)(ws + 3 * TENS);       // [B,T]
  int* delays = (int*)(ws + 3 * TENS + (size_t)B_ * T_ * sizeof(float));
  float* attn = q_t;                           // FFT writes in place per channel
  unsigned short* xbf = (unsigned short*)k_t;  // k_t dead after FFT kernel

  dim3 gg((B_ * T_) / 128, D_ / 128), bb(256);
  gemm_bias<float, true ><<<gg, bb, 0, stream>>>(query,  Wq, bq, q_t);
  gemm_bias<float, true ><<<gg, bb, 0, stream>>>(key_in, Wk, bk, k_t);
  gemm_bias<float, false><<<gg, bb, 0, stream>>>(value,  Wv, bv, v_);

  fft_corr<<<dim3(B_ * D_), bb, 0, stream>>>(q_t, k_t, attn);
  mean_kernel<<<dim3(T_ / 256, B_), bb, 0, stream>>>(attn, mean);
  topk_kernel<<<dim3(B_), bb, 0, stream>>>(mean, delays);
  softroll_kernel<<<dim3(T_ / 32, B_), bb, 0, stream>>>(attn, v_, delays, xbf);

  gemm_bias<unsigned short, false><<<gg, bb, 0, stream>>>(xbf, Wo, bo, out);
}
